// vSU2_CNN_1D_ref_41979010351317
// MI455X (gfx1250) — matrix operand order repaired and checked by reading, not compiled, not run on hardware
//
#include <hip/hip_runtime.h>
#include <hip/hip_bf16.h>
#include <math.h>

typedef __attribute__((ext_vector_type(16))) _Float16 v16h;
typedef __attribute__((ext_vector_type(8)))  _Float16 v8h;
typedef __attribute__((ext_vector_type(8)))  float    v8f;

#define NV 512   // n
#define MV 510   // m = n - 2

// Tetrahedron vertices (same constants as the reference, f32).
__constant__ float c_t[4][3] = {
  { 0.0f,                  0.0f,                  1.0f                 },
  { 0.94280904158206336f,  0.0f,                 -0.33333333333333331f },
  {-0.47140452079103168f,  0.81649658092772603f, -0.33333333333333331f },
  {-0.47140452079103168f, -0.81649658092772603f, -0.33333333333333331f }
};

// ---------------------------------------------------------------------------
// Kernel 1: build f16 operands and zero the y accumulator.
//   A16[f][j][k] = IK[f][j][k] - IK[f][k][j]   (zero-padded 510 -> 512)
//   D16[x][i][k] = t[ xv[(i+1+k)%512] ][x]     (zero-padded k >= 510)
// ---------------------------------------------------------------------------
__global__ void __launch_bounds__(256)
prep_kernel(const int* __restrict__ xv, const float* __restrict__ ik,
            _Float16* __restrict__ A16, _Float16* __restrict__ D16,
            float* __restrict__ yws)
{
  int idx = blockIdx.x * blockDim.x + threadIdx.x;   // [0, 3*512*512)
  int f = idx >> 18;
  int r = idx & 262143;
  int j = r >> 9;
  int k = r & 511;

  float av = 0.0f;
  if (j < MV && k < MV)
    av = ik[f * MV * MV + j * MV + k] - ik[f * MV * MV + k * MV + j];
  A16[idx] = (_Float16)av;

  float dv = 0.0f;
  if (k < MV) {
    int src = xv[(j + 1 + k) & (NV - 1)];   // here j plays the role of i
    dv = c_t[src][f];                       // here f plays the role of x
  }
  D16[idx] = (_Float16)dv;

  if (idx < NV * 4) yws[idx] = 0.0f;
}

// ---------------------------------------------------------------------------
// Kernel 2: one wave per 16x16 tile of G[x,f] = D_x @ A_f^T, fused epilogue
//   y[i,f] += sum_j cross(t[x[i]], t[x[(i+1+j)%512]])[x] * G[x,f][i,j]
// ---------------------------------------------------------------------------
__global__ void __launch_bounds__(32)
gemm_wmma_kernel(const int* __restrict__ xv,
                 const _Float16* __restrict__ A16,
                 const _Float16* __restrict__ D16,
                 float* __restrict__ yws)
{
  const int fx   = blockIdx.x;        // 0..8
  const int f    = fx / 3;
  const int xc   = fx % 3;
  const int i0   = blockIdx.y * 16;
  const int j0   = blockIdx.z * 16;
  const int lane = threadIdx.x;       // wave32
  const int half16 = lane >> 4;       // 0: lanes 0-15, 1: lanes 16-31
  const int l15    = lane & 15;

  const _Float16* Dbase = D16 + (size_t)xc * NV * NV;
  const _Float16* Bbase = A16 + (size_t)f  * NV * NV;

  // A-operand (16x32 f16): lane holds row (l15), 16 consecutive K (contiguous order, same as B).
  const int arow = i0 + l15;
  const int ak0  = half16 * 16;       // K base within 32-chunk: 0 or 16
  // B-operand (32x16 f16): lane holds column j, 16 consecutive K.
  const int bj   = j0 + l15;
  const int bk0  = half16 * 16;       // K base within 32-chunk: 0 or 16

  const _Float16* aptr = Dbase + (size_t)arow * NV + ak0;
  const _Float16* bptr = Bbase + (size_t)bj   * NV + bk0;

  v8f acc = {};
  #pragma unroll 4
  for (int kk = 0; kk < NV; kk += 32) {
    v8h alo = *(const v8h*)(aptr + kk);        // K = kk+ak0 .. +8
    v8h ahi = *(const v8h*)(aptr + kk + 8);    // K = kk+ak0+8 .. +8
    v8h blo = *(const v8h*)(bptr + kk);        // K = kk+bk0 .. +8
    v8h bhi = *(const v8h*)(bptr + kk + 8);    // K = kk+bk0+8 .. +8
    if (kk + 32 < NV) {
      __builtin_prefetch(aptr + kk + 32, 0, 1);
      __builtin_prefetch(bptr + kk + 32, 0, 1);
    }
    v16h a = __builtin_shufflevector(alo, ahi, 0,1,2,3,4,5,6,7,8,9,10,11,12,13,14,15);
    v16h b = __builtin_shufflevector(blo, bhi, 0,1,2,3,4,5,6,7,8,9,10,11,12,13,14,15);
    acc = __builtin_amdgcn_wmma_f32_16x16x32_f16(false, a, false, b,
                                                 (short)0, acc, false, false);
  }

  // Epilogue: multiply by C[i,j,x] and reduce over the 16 j's of this tile.
  union { v8f v; float f_[8]; } au; au.v = acc;
  const int j = bj;
  float s[8];
  if (j < MV) {
    #pragma unroll
    for (int r = 0; r < 8; ++r) {
      int i  = i0 + half16 * 8 + r;           // C/D layout: rows 0-7 / 8-15
      int ai = xv[i];
      int bi = xv[(i + 1 + j) & (NV - 1)];
      const float* u = c_t[ai];
      const float* v = c_t[bi];
      float cr;
      if (xc == 0)      cr = u[1] * v[2] - u[2] * v[1];
      else if (xc == 1) cr = u[2] * v[0] - u[0] * v[2];
      else              cr = u[0] * v[1] - u[1] * v[0];
      s[r] = au.f_[r] * cr;
    }
  } else {
    #pragma unroll
    for (int r = 0; r < 8; ++r) s[r] = 0.0f;
  }

  // Reduce across each 16-lane half (columns j0..j0+15).
  #pragma unroll
  for (int off = 1; off < 16; off <<= 1) {
    #pragma unroll
    for (int r = 0; r < 8; ++r) s[r] += __shfl_xor(s[r], off, 32);
  }

  if (l15 == 0) {
    #pragma unroll
    for (int r = 0; r < 8; ++r) {
      int i = i0 + half16 * 8 + r;
      atomicAdd(&yws[i * 4 + f], s[r]);
    }
  }
}

// ---------------------------------------------------------------------------
// Kernel 3: K=5 circular conv + bias + ELU + mean -> scalar
// ---------------------------------------------------------------------------
__global__ void __launch_bounds__(512)
finalize_kernel(const float* __restrict__ yws, const float* __restrict__ k0,
                const float* __restrict__ bias, float* __restrict__ out)
{
  __shared__ float ysh[NV][3];
  __shared__ float red[NV];
  int i = threadIdx.x;
  ysh[i][0] = yws[i * 4 + 0];
  ysh[i][1] = yws[i * 4 + 1];
  ysh[i][2] = yws[i * 4 + 2];
  __syncthreads();

  float zsum = 0.0f;
  #pragma unroll
  for (int fo = 0; fo < 3; ++fo) {
    float z = bias[fo];
    #pragma unroll
    for (int p = 0; p < 5; ++p) {
      int row = (p - i) & (NV - 1);           // (p - i) mod 512
      #pragma unroll
      for (int g = 0; g < 3; ++g)
        z += k0[(p * 3 + g) * 3 + fo] * ysh[row][g];
    }
    zsum += (z > 0.0f) ? z : (expf(z) - 1.0f);   // ELU(alpha=1)
  }
  red[i] = zsum;
  __syncthreads();
  for (int stride = 256; stride > 0; stride >>= 1) {
    if (i < stride) red[i] += red[i + stride];
    __syncthreads();
  }
  if (i == 0) out[0] = 0.1f + red[0] / 1536.0f;
}

// ---------------------------------------------------------------------------
extern "C" void kernel_launch(void* const* d_in, const int* in_sizes, int n_in,
                              void* d_out, int out_size, void* d_ws, size_t ws_size,
                              hipStream_t stream)
{
  const int*   xv   = (const int*)d_in[0];    // (512,) int32
  const float* ik   = (const float*)d_in[1];  // (3,510,510) f32
  const float* k0   = (const float*)d_in[2];  // (5,3,3) f32
  const float* bias = (const float*)d_in[3];  // (1,3) f32
  // d_in[4] = triple_product table: recomputed on device via cross products.
  float* out = (float*)d_out;

  char* ws = (char*)d_ws;
  _Float16* A16 = (_Float16*)ws;                               // 1.5 MB
  _Float16* D16 = (_Float16*)(ws + (size_t)3 * NV * NV * 2);   // 1.5 MB
  float*    yws = (float*)   (ws + (size_t)6 * NV * NV * 2);   // 8 KB

  prep_kernel<<<dim3((3 * NV * NV) / 256), dim3(256), 0, stream>>>(xv, ik, A16, D16, yws);
  gemm_wmma_kernel<<<dim3(9, NV / 16, NV / 16), dim3(32), 0, stream>>>(xv, A16, D16, yws);
  finalize_kernel<<<dim3(1), dim3(NV), 0, stream>>>(yws, k0, bias, out);
}
